// HausdorffDTLoss_56925496541913
// MI455X (gfx1250) — compile-verified, matrix operand order checked by reading, not run on hardware
//
#include <hip/hip_runtime.h>
#include <hip/hip_bf16.h>
#include <stdint.h>

typedef __attribute__((ext_vector_type(16))) _Float16 v16h;
typedef __attribute__((ext_vector_type(8)))  float    v8f;
typedef unsigned int u32x4 __attribute__((ext_vector_type(4)));
typedef int          i32x8 __attribute__((ext_vector_type(8)));
typedef int          i32x4 __attribute__((ext_vector_type(4)));

#define BB 4
#define CC 2
#define HH 256
#define WW 256
#define NP (BB*CC)          /* 8 images                          */
#define HW (HH*WW)          /* 65536                             */
#define NPHW (NP*HW)        /* 524288 elements                   */
#define NPLANES 32          /* (pred_fg,pred_bg,targ_fg,targ_bg) x 8 */
#define DT_INF 1.0e12f

__device__ __forceinline__ float sigmoid_f(float x) {
    return 1.0f / (1.0f + expf(-x));
}

// ---------------------------------------------------------------------------
// 1) Fused mask-init + 1D min-plus DT along W.
//    Block = 256 threads = 4 rows of one plane; thread computes 4 outputs
//    (i, i+64, i+128, i+192) so one LDS broadcast read feeds 12 VALU ops.
//    Row index is wave-uniform -> conflict-free LDS broadcast.
// ---------------------------------------------------------------------------
__global__ void hd_dt_w_fused(const float* __restrict__ preds,
                              const float* __restrict__ targets,
                              float* __restrict__ dt) {
    __shared__ float f[4][WW];
    const int blk   = blockIdx.x;
    const int plane = blk >> 6;            // / (HH/4)
    const int r0    = (blk & 63) * 4;
    const int kind  = plane >> 3;          // 0:pred_fg 1:pred_bg 2:targ_fg 3:targ_bg
    const int img   = plane & 7;
    const int t     = threadIdx.x;
    const int rr    = t >> 6;              // wave-uniform
    const int c4    = (t & 63) * 4;
    const int row   = r0 + rr;

    const float* src = (kind < 2) ? preds : targets;
    const float  thr = (kind < 2) ? 0.0f : 0.5f;   // sigmoid(x)>0.5 <=> x>0
    const bool   bg  = (kind & 1);

    const float* sp = src + (size_t)img * HW + (size_t)row * WW + c4;
    __builtin_prefetch(sp, 0, 1);                  // global_prefetch_b8
    const float4 v = *(const float4*)sp;
    f[rr][c4 + 0] = ((v.x > thr) != bg) ? DT_INF : 0.0f;
    f[rr][c4 + 1] = ((v.y > thr) != bg) ? DT_INF : 0.0f;
    f[rr][c4 + 2] = ((v.z > thr) != bg) ? DT_INF : 0.0f;
    f[rr][c4 + 3] = ((v.w > thr) != bg) ? DT_INF : 0.0f;
    __syncthreads();

    const float base = (float)(t & 63);
    float m0 = DT_INF, m1 = DT_INF, m2 = DT_INF, m3 = DT_INF;
    #pragma unroll 4
    for (int j = 0; j < WW; ++j) {
        const float fj = f[rr][j];
        const float d0 = base - (float)j;
        const float d1 = d0 + 64.0f, d2 = d0 + 128.0f, d3 = d0 + 192.0f;
        m0 = fminf(m0, fmaf(d0, d0, fj));
        m1 = fminf(m1, fmaf(d1, d1, fj));
        m2 = fminf(m2, fmaf(d2, d2, fj));
        m3 = fminf(m3, fmaf(d3, d3, fj));
    }
    float* dst = dt + (size_t)plane * HW + (size_t)row * WW;
    const int i0 = t & 63;
    dst[i0      ] = m0;
    dst[i0 +  64] = m1;
    dst[i0 + 128] = m2;
    dst[i0 + 192] = m3;
}

// ---------------------------------------------------------------------------
// 2) DT along H. The strided column gather (4 cols x 256 rows, row stride
//    256 floats) is done by the Tensor Data Mover as one 2D-tile DMA into
//    LDS (tensor_load_to_lds + s_wait_tensorcnt), issued by wave 0.
//    LDS tile layout (row-major 256x4) matches TDM's X-fastest fill order.
//    Fuses the final sqrt(min(f, INF)).
// ---------------------------------------------------------------------------
__global__ void hd_dt_h_tdm(float* __restrict__ dt) {
    __shared__ float ft[HH][4];
    const int blk   = blockIdx.x;
    const int plane = blk >> 6;            // / (WW/4)
    const int c0    = (blk & 63) * 4;
    const int t     = threadIdx.x;

    float* gsrc = dt + (size_t)plane * HW + c0;

    if (t < 32) {
        const unsigned long long ga = (unsigned long long)(uintptr_t)gsrc;
        const unsigned int lds_off  = (unsigned int)(uintptr_t)&ft[0][0];
        // D# group0: count=1 | lds_addr | global_addr[56:0] | type=2
        u32x4 g0 = { 1u,
                     lds_off,
                     (unsigned int)(ga & 0xFFFFFFFFull),
                     (unsigned int)((ga >> 32) & 0x01FFFFFFull) | 0x80000000u };
        // D# group1: data_size=4B; tensor 256x256; tile 4x256; stride0=256,
        //            stride1=65536 (elements)
        i32x8 g1 = { 0x00020000,        // wg_mask=0, data_size=2 (4B)
                     0x01000000,        // tensor_dim0 lo16 = 256
                     0x01000000,        // tensor_dim0 hi16=0 | tensor_dim1 lo16 = 256
                     0x00040000,        // tensor_dim1 hi16=0 | tile_dim0 = 4
                     0x00000100,        // tile_dim1 = 256, tile_dim2 = 0
                     0x00000100,        // tensor_dim0_stride = 256 (lo32)
                     0x00000000,        // stride0 hi16 | stride1 lo16 (= 0)
                     0x00000001 };      // tensor_dim1_stride[47:16] = 1 -> 65536
        i32x4 gz = { 0, 0, 0, 0 };
#if defined(__clang_major__) && (__clang_major__ >= 23)
        i32x8 gz8 = { 0, 0, 0, 0, 0, 0, 0, 0 };
        __builtin_amdgcn_tensor_load_to_lds(g0, g1, gz, gz, gz8, 0);
#else
        __builtin_amdgcn_tensor_load_to_lds(g0, g1, gz, gz, 0);
#endif
        __builtin_amdgcn_s_wait_tensorcnt(0);
    }
    __syncthreads();

    const int cc = t >> 6;                 // wave-uniform column
    const float base = (float)(t & 63);
    float m0 = DT_INF, m1 = DT_INF, m2 = DT_INF, m3 = DT_INF;
    #pragma unroll 4
    for (int j = 0; j < HH; ++j) {
        const float fj = ft[j][cc];        // wave-uniform -> broadcast
        const float d0 = base - (float)j;
        const float d1 = d0 + 64.0f, d2 = d0 + 128.0f, d3 = d0 + 192.0f;
        m0 = fminf(m0, fmaf(d0, d0, fj));
        m1 = fminf(m1, fmaf(d1, d1, fj));
        m2 = fminf(m2, fmaf(d2, d2, fj));
        m3 = fminf(m3, fmaf(d3, d3, fj));
    }
    float* dcol = dt + (size_t)plane * HW + c0 + cc;
    const int i0 = t & 63;
    dcol[(size_t)(i0      ) * WW] = sqrtf(fminf(m0, DT_INF));
    dcol[(size_t)(i0 +  64) * WW] = sqrtf(fminf(m1, DT_INF));
    dcol[(size_t)(i0 + 128) * WW] = sqrtf(fminf(m2, DT_INF));
    dcol[(size_t)(i0 + 192) * WW] = sqrtf(fminf(m3, DT_INF));
}

// ---------------------------------------------------------------------------
// 3) Per-plane max of the (already sqrt'ed) distance field; read-only.
//    any(mask) <=> max(dist) > 0, so the max doubles as the any-flag.
// ---------------------------------------------------------------------------
__global__ void hd_plane_max(const float* __restrict__ dt,
                             float* __restrict__ stats) {
    __shared__ float smax[256];
    const int plane = blockIdx.x;
    const int t     = threadIdx.x;
    const float* p = dt + (size_t)plane * HW;
    float mv = 0.0f;
    for (int i = t; i < HW; i += 256) mv = fmaxf(mv, p[i]);
    smax[t] = mv;
    __syncthreads();
    for (int s = 128; s > 0; s >>= 1) {
        if (t < s) smax[t] = fmaxf(smax[t], smax[t + s]);
        __syncthreads();
    }
    if (t == 0) stats[plane] = smax[0];
}

__global__ void hd_zero_out(float* __restrict__ out) { out[0] = 0.0f; }

// ---------------------------------------------------------------------------
// 4) Final reduction mean(err * distance) as D = A x ones + C on the matrix
//    core (v_wmma_f32_16x16x32_f16, f32 accumulator); each wave ingests 512
//    values per WMMA. Uniform control flow => EXEC all-ones as required.
// ---------------------------------------------------------------------------
__global__ void hd_loss_wmma(const float* __restrict__ preds,
                             const float* __restrict__ targets,
                             const float* __restrict__ dt,
                             const float* __restrict__ stats,
                             float* __restrict__ out) {
    const int g = blockIdx.x * blockDim.x + threadIdx.x;   // 0..32767
    const int G = 32768;

    v16h a;
    v16h ones;
    #pragma unroll
    for (int k = 0; k < 16; ++k) ones[k] = (_Float16)1.0f;
    v8f acc = {};

    #pragma unroll
    for (int k = 0; k < 16; ++k) {
        const int e   = k * G + g;         // coalesced per slot
        const int img = e >> 16;
        const float maxfgp = stats[img];
        const float maxbgp = stats[img +  8];
        const float maxfgt = stats[img + 16];
        const float maxbgt = stats[img + 24];
        const float fgp = dt[e];
        const float bgp = dt[e +  8 * HW];
        const float fgt = dt[e + 16 * HW];
        const float bgt = dt[e + 24 * HW];
        const float fieldp = (maxfgp > 0.0f)
            ? (fgp / fmaxf(maxfgp, 1e-12f) +
               ((maxbgp > 0.0f) ? bgp / fmaxf(maxbgp, 1e-12f) : 0.0f))
            : 0.0f;
        const float fieldt = (maxfgt > 0.0f)
            ? (fgt / fmaxf(maxfgt, 1e-12f) +
               ((maxbgt > 0.0f) ? bgt / fmaxf(maxbgt, 1e-12f) : 0.0f))
            : 0.0f;
        const float distance = fieldp * fieldp + fieldt * fieldt;  // ALPHA=2
        const float pv  = sigmoid_f(preds[e]);
        float err = pv - targets[e];
        err *= err;
        a[k] = (_Float16)(err * distance);
    }

    acc = __builtin_amdgcn_wmma_f32_16x16x32_f16(
        /*neg_a=*/false, a, /*neg_b=*/false, ones,
        /*c_mod=*/(short)0, acc, /*reuse_a=*/false, /*reuse_b=*/false);

    float s = acc[0] + acc[1] + acc[2] + acc[3] +
              acc[4] + acc[5] + acc[6] + acc[7];
    #pragma unroll
    for (int off = 16; off > 0; off >>= 1) s += __shfl_xor(s, off, 32);

    if ((threadIdx.x & 31) == 0) {
        atomicAdd(out, s * (1.0f / 16.0f) * (1.0f / (float)NPHW));
    }
}

// ---------------------------------------------------------------------------
extern "C" void kernel_launch(void* const* d_in, const int* in_sizes, int n_in,
                              void* d_out, int out_size, void* d_ws, size_t ws_size,
                              hipStream_t stream) {
    const float* preds   = (const float*)d_in[0];
    const float* targets = (const float*)d_in[1];
    // d_in[2] = labels: unused by the reference loss.
    float* out = (float*)d_out;

    float* dt    = (float*)d_ws;               // 32 * 65536 floats = 8 MiB
    float* stats = dt + (size_t)NPLANES * HW;  // 32 floats

    hd_dt_w_fused<<<NPLANES * (HH / 4), 256, 0, stream>>>(preds, targets, dt);
    hd_dt_h_tdm  <<<NPLANES * (WW / 4), 256, 0, stream>>>(dt);
    hd_plane_max <<<NPLANES, 256, 0, stream>>>(dt, stats);
    hd_zero_out  <<<1, 1, 0, stream>>>(out);
    hd_loss_wmma <<<128, 256, 0, stream>>>(preds, targets, dt, stats, out);
}
